// GRUModel_5970004541474
// MI455X (gfx1250) — compile-verified, matrix operand order checked by reading, not run on hardware
//
#include <hip/hip_runtime.h>

// ---- CDNA5 WMMA types ------------------------------------------------------
typedef __attribute__((ext_vector_type(16))) _Float16 v16h;
typedef __attribute__((ext_vector_type(8)))  _Float16 v8h;
typedef __attribute__((ext_vector_type(8)))  float    v8f;

#define HID   32            // GRU hidden size (== WMMA K !)
#define VOC   95            // input/output vocab
#define SEQT  256           // timesteps
#define KPAD  96            // layer-1 K padded to 3*32
#define NG    96            // 3*H gate width

// Fragment table in LDS (f16, pre-swizzled B-operand layout, 1KB each):
//  [ 0..17] Wih1  (kt*6+nt)   [18..23] Whh1   [24..29] Wih2
//  [30..35] Whh2  [36..41] Wih3   [42..47] Whh3   [48..53] Wfc
#define NFRAG 54

static __device__ __forceinline__ v8f wmma_f16(v16h a, v16h b, v8f c) {
  return __builtin_amdgcn_wmma_f32_16x16x32_f16(
      /*neg_a=*/false, a, /*neg_b=*/false, b,
      /*c_mod=*/(short)0, c, /*reuse_a=*/false, /*reuse_b=*/false);
}

static __device__ __forceinline__ v8f zero8() {
  v8f r;
#pragma unroll
  for (int i = 0; i < 8; ++i) r[i] = 0.0f;
  return r;                       // folds to inline SRC2 = 0 in v_wmma
}

static __device__ __forceinline__ float fsigmoid(float x) {
  return __builtin_amdgcn_rcpf(1.0f + __expf(-x));
}
static __device__ __forceinline__ float ftanh(float x) {
  return 2.0f * fsigmoid(2.0f * x) - 1.0f;   // robust at +/-inf
}

// Build one B-operand fragment for D = A x B with B[k][n] = W[n][k].
// Lane layout mirrors the 16-bit operand layout of ISA 7.12.2.
static __device__ __forceinline__ void build_frag(
    _Float16* __restrict__ dst /*[32][16]*/, const float* __restrict__ W,
    int ldW, int n0, int k0, int Nvalid, int Kvalid, int lane) {
  const int n  = n0 + (lane & 15);
  const int hi = lane >> 4;
#pragma unroll
  for (int v = 0; v < 8; ++v) {
#pragma unroll
    for (int j = 0; j < 2; ++j) {
      const int kl = (v & 3) * 2 + j + hi * 8 + (v >> 2) * 16;
      const int k  = k0 + kl;
      float val = (n < Nvalid && k < Kvalid) ? W[(size_t)n * ldW + k] : 0.0f;
      dst[lane * 16 + v * 2 + j] = (_Float16)val;
    }
  }
}

static __device__ __forceinline__ v16h load_wfrag(const _Float16* f, int lane) {
  return *(const v16h*)(f + lane * 16);          // 2x ds_load_b128
}

// Load A operand (16 x 32 f16, row-major [rows][ldk] in LDS) for K-tile kt.
static __device__ __forceinline__ v16h load_A(const _Float16* base, int ldk,
                                              int kt, int lane) {
  const int m  = lane & 15;
  const int hi = lane >> 4;
  const _Float16* row = base + m * ldk + kt * 32 + hi * 8;
  v8h a = *(const v8h*)(row);        // k_local = hi*8 .. hi*8+7
  v8h b = *(const v8h*)(row + 16);   // k_local = 16+hi*8 .. +7
  v16h r;
#pragma unroll
  for (int i = 0; i < 8; ++i) { r[i] = a[i]; r[8 + i] = b[i]; }
  return r;
}

// Scatter this wave's 16x16 f32 half-tile into a 16x32 staged h (f16 A-source).
static __device__ __forceinline__ void store_half(_Float16* dst /*[16][32]*/,
                                                  v8f h, int lane, int wid) {
  const int u  = wid * 16 + (lane & 15);
  const int hi = lane >> 4;
#pragma unroll
  for (int r = 0; r < 8; ++r)
    dst[(r + 8 * hi) * HID + u] = (_Float16)h[r];
}

// GRU update for this wave's 16 hidden units.  WMMA C started at 0, so the
// biases come in here: br = bih_r+bhh_r, bz = bih_z+bhh_z; the n-gate keeps
// bhh_n INSIDE the r*(.) term per PyTorch GRU semantics.
static __device__ __forceinline__ v8f gru_update1(v8f h, const v8f* ip,
                                                  const v8f* hp, float br,
                                                  float bz, float bin,
                                                  float bhn) {
  v8f hn;
#pragma unroll
  for (int r = 0; r < 8; ++r) {
    float rg = fsigmoid(ip[0][r] + hp[0][r] + br);
    float zg = fsigmoid(ip[1][r] + hp[1][r] + bz);
    float ng = ftanh  (ip[2][r] + bin + rg * (hp[2][r] + bhn));
    hn[r] = ng + zg * (h[r] - ng);
  }
  return hn;
}

__global__ __launch_bounds__(64) void gru3_fused_wmma(
    const float* __restrict__ x,
    const float* __restrict__ Wih1, const float* __restrict__ Whh1,
    const float* __restrict__ bih1, const float* __restrict__ bhh1,
    const float* __restrict__ Wih2, const float* __restrict__ Whh2,
    const float* __restrict__ bih2, const float* __restrict__ bhh2,
    const float* __restrict__ Wih3, const float* __restrict__ Whh3,
    const float* __restrict__ bih3, const float* __restrict__ bhh3,
    const float* __restrict__ Wfc,  const float* __restrict__ bfc,
    float* __restrict__ out, int B, int T) {
  __shared__ alignas(32) _Float16 wfrag[NFRAG][32][16];   // 54 KB
  __shared__ alignas(32) _Float16 xstage[16][KPAD];       // 3 KB  rd [A,B) wr [C,A)
  __shared__ alignas(32) _Float16 h1buf[16][HID];         // 1 KB  rd [A,B) wr [C,A)
  __shared__ alignas(32) _Float16 hout[16][HID];          // 1 KB  rd [A,B) wr [C,A)
  __shared__ alignas(32) _Float16 hin2[16][HID];          // 1 KB  rd [B,C) wr [A,B)
  __shared__ alignas(32) _Float16 h2buf[16][HID];         // 1 KB  rd [B,C) wr [A,B)
  __shared__ alignas(32) _Float16 hin3[16][HID];          // 1 KB  rd [C,A) wr [B,C)
  __shared__ alignas(32) _Float16 h3buf[16][HID];         // 1 KB  rd [C,A) wr [B,C)

  const int tid  = threadIdx.x;
  const int lane = tid & 31;
  const int wid  = tid >> 5;             // wave 0 / wave 1: hidden-unit half
  const int b0   = blockIdx.x * 16;      // 16 batch rows per workgroup
  if (b0 >= B) return;

  // ---- one-time: swizzle weights into LDS B-fragments, split by parity ----
#pragma unroll
  for (int f = 0; f < 18; ++f)
    if ((f & 1) == wid)
      build_frag(&wfrag[f][0][0], Wih1, VOC, (f % 6) * 16, (f / 6) * 32, NG, VOC, lane);
#pragma unroll
  for (int nt = 0; nt < 6; ++nt) {
    if (((18 + nt) & 1) == wid) build_frag(&wfrag[18 + nt][0][0], Whh1, HID, nt * 16, 0, NG,  HID, lane);
    if (((24 + nt) & 1) == wid) build_frag(&wfrag[24 + nt][0][0], Wih2, HID, nt * 16, 0, NG,  HID, lane);
    if (((30 + nt) & 1) == wid) build_frag(&wfrag[30 + nt][0][0], Whh2, HID, nt * 16, 0, NG,  HID, lane);
    if (((36 + nt) & 1) == wid) build_frag(&wfrag[36 + nt][0][0], Wih3, HID, nt * 16, 0, NG,  HID, lane);
    if (((42 + nt) & 1) == wid) build_frag(&wfrag[42 + nt][0][0], Whh3, HID, nt * 16, 0, NG,  HID, lane);
    if (((48 + nt) & 1) == wid) build_frag(&wfrag[48 + nt][0][0], Wfc,  HID, nt * 16, 0, VOC, HID, lane);
  }

  // ---- this wave owns gate ntiles {wid, 2+wid, 4+wid} ----
  const int col = lane & 15;
  const int hi  = lane >> 4;
  int nts[3];
#pragma unroll
  for (int i = 0; i < 3; ++i) nts[i] = wid + 2 * i;

  // combined bias scalars (per lane column)
  float br1, bz1, bin1, bhn1, br2, bz2, bin2, bhn2, br3, bz3, bin3, bhn3, bo[3];
  {
    const int nr = nts[0] * 16 + col, nz = nts[1] * 16 + col, nn = nts[2] * 16 + col;
    br1 = bih1[nr] + bhh1[nr]; bz1 = bih1[nz] + bhh1[nz]; bin1 = bih1[nn]; bhn1 = bhh1[nn];
    br2 = bih2[nr] + bhh2[nr]; bz2 = bih2[nz] + bhh2[nz]; bin2 = bih2[nn]; bhn2 = bhh2[nn];
    br3 = bih3[nr] + bhh3[nr]; bz3 = bih3[nz] + bhh3[nz]; bin3 = bih3[nn]; bhn3 = bhh3[nn];
#pragma unroll
    for (int i = 0; i < 3; ++i) {
      const int n = nts[i] * 16 + col;
      bo[i] = (n < VOC) ? bfc[n] : 0.0f;
    }
  }

  // hidden-state halves (16 batch rows x 16 units) in registers, h0 = 0
  v8f h1 = zero8(), h2 = zero8(), h3 = zero8();

  // ---- prologue: stage x(t=0) (with zero pad col 95) + zero h1buf/hout ----
#pragma unroll
  for (int rr = 0; rr < 8; ++rr) {
    const int row = wid * 8 + rr;
    const float* xr = x + (size_t)(b0 + row) * T * VOC;
#pragma unroll
    for (int c = 0; c < 3; ++c) {
      const int cc = lane + 32 * c;
      xstage[row][cc] = (cc < VOC) ? (_Float16)xr[cc] : (_Float16)0.0f;
    }
  }
  store_half(&h1buf[0][0], h1, lane, wid);
  store_half(&hout[0][0],  h3, lane, wid);

  for (int t = 0; t < T; ++t) {
    __syncthreads();                                       // barrier A
    // ---- segment [A,B): FC of t-1 + full layer 1 (15 independent WMMAs) ----
    v16h ax0 = load_A(&xstage[0][0], KPAD, 0, lane);
    v16h ax1 = load_A(&xstage[0][0], KPAD, 1, lane);
    v16h ax2 = load_A(&xstage[0][0], KPAD, 2, lane);
    v16h ah1 = load_A(&h1buf[0][0],  HID,  0, lane);
    v16h aho = load_A(&hout[0][0],   HID,  0, lane);

    v8f o[3], ip[3], hp[3];
#pragma unroll
    for (int i = 0; i < 3; ++i)
      o[i] = wmma_f16(aho, load_wfrag(&wfrag[48 + nts[i]][0][0], lane), zero8());
#pragma unroll
    for (int i = 0; i < 3; ++i) {
      v8f c = zero8();
      c = wmma_f16(ax0, load_wfrag(&wfrag[0 + nts[i]][0][0],  lane), c);
      c = wmma_f16(ax1, load_wfrag(&wfrag[6 + nts[i]][0][0],  lane), c);
      c = wmma_f16(ax2, load_wfrag(&wfrag[12 + nts[i]][0][0], lane), c);
      ip[i] = c;
      hp[i] = wmma_f16(ah1, load_wfrag(&wfrag[18 + nts[i]][0][0], lane), zero8());
    }

    // emit output of step t-1 (uniform guard; all WMMAs above already issued)
    if (t > 0) {
#pragma unroll
      for (int i = 0; i < 3; ++i) {
        const int n = nts[i] * 16 + col;
        if (n < VOC) {
#pragma unroll
          for (int r = 0; r < 8; ++r)
            out[((size_t)(b0 + r + 8 * hi) * T + (t - 1)) * VOC + n] = o[i][r] + bo[i];
        }
      }
    }

    // prefetch x(t+1) into registers (latency hidden behind L1..L3 compute)
    float xn[8][3];
    if (t + 1 < T) {
#pragma unroll
      for (int rr = 0; rr < 8; ++rr) {
        const int row = wid * 8 + rr;
        const float* xr = x + ((size_t)(b0 + row) * T + (t + 1)) * VOC;
        __builtin_prefetch(xr + VOC, 0, 1);                // warm t+2
#pragma unroll
        for (int c = 0; c < 3; ++c) {
          const int cc = lane + 32 * c;
          xn[rr][c] = (cc < VOC) ? xr[cc] : 0.0f;
        }
      }
    }

    h1 = gru_update1(h1, ip, hp, br1, bz1, bin1, bhn1);
    store_half(&hin2[0][0],  h1, lane, wid);               // new h1 = L2 input
    store_half(&h2buf[0][0], h2, lane, wid);               // old h2 = L2 state
    __syncthreads();                                       // barrier B

    // ---- segment [B,C): layer 2 (6 WMMAs) ----
    {
      v16h a0 = load_A(&hin2[0][0],  HID, 0, lane);
      v16h a1 = load_A(&h2buf[0][0], HID, 0, lane);
#pragma unroll
      for (int i = 0; i < 3; ++i) {
        ip[i] = wmma_f16(a0, load_wfrag(&wfrag[24 + nts[i]][0][0], lane), zero8());
        hp[i] = wmma_f16(a1, load_wfrag(&wfrag[30 + nts[i]][0][0], lane), zero8());
      }
      h2 = gru_update1(h2, ip, hp, br2, bz2, bin2, bhn2);
    }
    store_half(&hin3[0][0],  h2, lane, wid);
    store_half(&h3buf[0][0], h3, lane, wid);
    __syncthreads();                                       // barrier C

    // ---- segment [C,A): layer 3 (6 WMMAs) + next-step staging ----
    {
      v16h a0 = load_A(&hin3[0][0],  HID, 0, lane);
      v16h a1 = load_A(&h3buf[0][0], HID, 0, lane);
#pragma unroll
      for (int i = 0; i < 3; ++i) {
        ip[i] = wmma_f16(a0, load_wfrag(&wfrag[36 + nts[i]][0][0], lane), zero8());
        hp[i] = wmma_f16(a1, load_wfrag(&wfrag[42 + nts[i]][0][0], lane), zero8());
      }
      h3 = gru_update1(h3, ip, hp, br3, bz3, bin3, bhn3);
    }
    store_half(&hout[0][0],  h3, lane, wid);               // FC input for t+1
    store_half(&h1buf[0][0], h1, lane, wid);               // L1 state for t+1
    if (t + 1 < T) {                                       // xstage for t+1
#pragma unroll
      for (int rr = 0; rr < 8; ++rr) {
        const int row = wid * 8 + rr;
#pragma unroll
        for (int c = 0; c < 3; ++c) {
          const int cc = lane + 32 * c;
          if (cc < VOC) xstage[row][cc] = (_Float16)xn[rr][c];
        }
      }
    }
  }

  // ---- epilogue: FC head for t = T-1 ----
  __syncthreads();
  {
    v16h aho = load_A(&hout[0][0], HID, 0, lane);
    v8f o[3];
#pragma unroll
    for (int i = 0; i < 3; ++i)
      o[i] = wmma_f16(aho, load_wfrag(&wfrag[48 + nts[i]][0][0], lane), zero8());
#pragma unroll
    for (int i = 0; i < 3; ++i) {
      const int n = nts[i] * 16 + col;
      if (n < VOC) {
#pragma unroll
        for (int r = 0; r < 8; ++r)
          out[((size_t)(b0 + r + 8 * hi) * T + (T - 1)) * VOC + n] = o[i][r] + bo[i];
      }
    }
  }
}

extern "C" void kernel_launch(void* const* d_in, const int* in_sizes, int n_in,
                              void* d_out, int out_size, void* d_ws, size_t ws_size,
                              hipStream_t stream) {
  (void)n_in; (void)d_ws; (void)ws_size; (void)out_size;
  const float* x    = (const float*)d_in[0];
  const float* Wih1 = (const float*)d_in[1];
  const float* Whh1 = (const float*)d_in[2];
  const float* bih1 = (const float*)d_in[3];
  const float* bhh1 = (const float*)d_in[4];
  const float* Wih2 = (const float*)d_in[5];
  const float* Whh2 = (const float*)d_in[6];
  const float* bih2 = (const float*)d_in[7];
  const float* bhh2 = (const float*)d_in[8];
  const float* Wih3 = (const float*)d_in[9];
  const float* Whh3 = (const float*)d_in[10];
  const float* bih3 = (const float*)d_in[11];
  const float* bhh3 = (const float*)d_in[12];
  const float* Wfc  = (const float*)d_in[13];
  const float* bfc  = (const float*)d_in[14];
  float* out = (float*)d_out;

  const int T = SEQT;
  const int B = in_sizes[0] / (T * VOC);   // 2048 for the reference setup
  const int nblocks = (B + 15) / 16;       // 2 waves (one 16-row tile) per block

  hipLaunchKernelGGL(gru3_fused_wmma, dim3(nblocks), dim3(64), 0, stream,
                     x, Wih1, Whh1, bih1, bhh1, Wih2, Whh2, bih2, bhh2,
                     Wih3, Whh3, bih3, bhh3, Wfc, bfc, out, B, T);
}
